// scflow_52338471469688
// MI455X (gfx1250) — matrix operand order repaired and checked by reading, not compiled, not run on hardware
//
#include <hip/hip_runtime.h>
#include <cstdint>

// ---------------------------------------------------------------------------
// Types for CDNA5 WMMA (wave32). 16x16x32 bf16 -> f32 accumulate.
// ---------------------------------------------------------------------------
typedef __attribute__((ext_vector_type(16))) __bf16 v16bf;
typedef __attribute__((ext_vector_type(8)))  __bf16 v8bf;
typedef __attribute__((ext_vector_type(8)))  float  v8f;
typedef __attribute__((ext_vector_type(4)))  int    v4i;

// ---------------------------------------------------------------------------
// gfx1250 async global->LDS copy (probe-confirmed in round 2/3: params are
// v4i pointers in AS1 / AS3). Falls back to plain stores if absent.
// ---------------------------------------------------------------------------
#if defined(__HIP_DEVICE_COMPILE__) && __has_builtin(__builtin_amdgcn_global_load_async_to_lds_b128)
#define HAVE_ASYNC_LDS 1
#else
#define HAVE_ASYNC_LDS 0
#endif

#if HAVE_ASYNC_LDS
typedef __attribute__((address_space(1))) v4i GlobV4;
typedef __attribute__((address_space(3))) v4i LdsV4;
__device__ __forceinline__ void async_copy_b128(const void* gsrc, void* ldst) {
  // generic->AS pointers via integer round-trip (low 32 bits of a generic LDS
  // address are the LDS byte offset per the gfx1250 aperture rules)
  GlobV4* gp = (GlobV4*)(unsigned long long)(uintptr_t)gsrc;
  LdsV4* lp = (LdsV4*)(unsigned)(uintptr_t)ldst;
  __builtin_amdgcn_global_load_async_to_lds_b128(gp, lp, 0, 0);
}
__device__ __forceinline__ void async_wait0() {
#if __has_builtin(__builtin_amdgcn_s_wait_asynccnt)
  __builtin_amdgcn_s_wait_asynccnt(0);
#else
  asm volatile("s_wait_asynccnt 0" ::: "memory");
#endif
}
#endif

// ---------------------------------------------------------------------------
// Bilinear sample with zero padding (grid_sample semantics of the reference)
// ---------------------------------------------------------------------------
__device__ __forceinline__ float bilerp_zero(const float* __restrict__ img, int H, int W,
                                             float gx, float gy) {
  float x0 = floorf(gx), y0 = floorf(gy);
  float wx = gx - x0, wy = gy - y0;
  float s = 0.f;
#pragma unroll
  for (int dy = 0; dy < 2; ++dy) {
#pragma unroll
    for (int dx = 0; dx < 2; ++dx) {
      float xi = x0 + (float)dx, yi = y0 + (float)dy;
      if (xi >= 0.f && xi <= (float)(W - 1) && yi >= 0.f && yi <= (float)(H - 1)) {
        float wgt = (dx ? wx : 1.f - wx) * (dy ? wy : 1.f - wy);
        s += wgt * img[(int)yi * W + (int)xi];
      }
    }
  }
  return s;
}

// ---------------------------------------------------------------------------
// One-time weight conversion: OIHW fp32 -> bf16 shadow, [Np=ceil64(Cout)] rows
// of [Kp=ceil32(K)] elements, zero padded in both dims. Makes the conv's B
// staging an unconditional aligned async b128 copy with no bounds logic, and
// removes all fp32->bf16 converts from the conv K-loop.
// ---------------------------------------------------------------------------
__global__ void w2bf_kernel(const float* __restrict__ w, __bf16* __restrict__ o,
                            int Cout, int K, int Kp, long total) {
  long i = (long)blockIdx.x * blockDim.x + threadIdx.x;
  if (i >= total) return;
  int n = (int)(i / Kp);
  int k = (int)(i - (long)n * Kp);
  float v = (n < Cout && k < K) ? w[(long)n * K + k] : 0.f;
  o[i] = (__bf16)v;
}

// ---------------------------------------------------------------------------
// LDS-staged implicit-GEMM conv (NCHW), v_wmma_f32_16x16x32_bf16.
// M = B*Hout*Wout, N = Cout, K = Cin*KS*KS; weights pre-converted to bf16
// with row stride Kp (zero padded).
// Block: 128 threads = 4 waves. Block tile: 16(M) x 64(N); wave w covers
// N columns [nBase+16w, nBase+16w+16). K loop in steps of 32.
// Per step: A tile (16x32 bf16, 1KB) staged via im2col gather (4 elem/thread);
// B tile (64x32 bf16, 4KB) staged via async global->LDS b128 (2 chunks/thread).
// Fragment layouts per CDNA5 ISA 7.12.2:
//   A (16x32 bf16): lane m = lane&15, g = lane>>4; elem e -> k=(e<8?e:e+8)+8g
//     => two contiguous 8-elem runs: [8g, 8g+8) and [16+8g, 16+8g+8)
//   B (32x16 bf16): lane n = lane&15; elem e -> k=(e<8?e:e+8)+8g  (same two runs as A)
//   C/D (16x16 f32): elem r -> m = r + 8g, n = lane&15
// ---------------------------------------------------------------------------
template <int KS>
__global__ __launch_bounds__(128) void conv_wmma_kernel(
    const float* __restrict__ x, long xBS,          // input, batch stride (elems)
    const __bf16* __restrict__ wbf, int Kp,         // bf16 weights, row stride Kp
    const float* __restrict__ bias,                 // (Cout)
    float* __restrict__ out, long oBS,              // output, batch stride (elems)
    int Bn, int Cin, int Hin, int Win,
    int Cout, int Hout, int Wout,
    int stride, int leaky) {
  __shared__ __bf16 As[16 * 32];   // 1 KB
  __shared__ __bf16 Bsb[64 * 32];  // 4 KB

  const int tid = threadIdx.x;
  const int wave = tid >> 5;
  const int lane = tid & 31;
  const int laneM = lane & 15;
  const int grp = lane >> 4;
  const int HWo = Hout * Wout;
  const int M = Bn * HWo;
  const int ks2 = KS * KS;
  const int K = Cin * ks2;
  const int pad = (KS - 1) / 2;
  const int mTile = blockIdx.x;
  const int nBase = blockIdx.y << 6;
  const int n = nBase + (wave << 4) + laneM;       // this lane's output channel

  // A-staging: this thread fills row ms, k slots [kloc0, kloc0+4)
  const int ms = tid >> 3;
  const int kloc0 = (tid & 7) << 2;
  const int mgS = mTile * 16 + ms;
  const bool sValid = (mgS < M);
  int sb = 0, soy = 0, sox = 0;
  if (sValid) {
    sb = mgS / HWo;
    int r0 = mgS - sb * HWo;
    soy = r0 / Wout;
    sox = r0 - soy * Wout;
  }
  const long sxbase = (long)sb * xBS;

  v8f acc = {};
  for (int kk = 0; kk < K; kk += 32) {
    // ---- stage A (im2col gather, fp32 -> bf16) ----
#pragma unroll
    for (int j = 0; j < 4; ++j) {
      int k = kk + kloc0 + j;
      float va = 0.f;
      if (sValid && k < K) {
        int ci = k / ks2;
        int t = k - ci * ks2;
        int ky = t / KS;
        int kx = t - ky * KS;
        int iy = soy * stride + ky - pad;
        int ix = sox * stride + kx - pad;
        if (iy >= 0 && iy < Hin && ix >= 0 && ix < Win)
          va = x[sxbase + ((long)ci * Hin + iy) * Win + ix];
      }
      As[ms * 32 + kloc0 + j] = (__bf16)va;
    }
    // ---- stage B (bf16 weights; shadow is padded & zero-filled so every
    //      address is valid and every padded element is a true zero) ----
#if HAVE_ASYNC_LDS
#pragma unroll
    for (int i = 0; i < 2; ++i) {
      int q = tid * 2 + i;                 // 16-byte chunk id, [0,256)
      int nr = q >> 2;                     // row in [0,64)
      int kp8 = (q & 3) << 3;              // bf16 offset in row's 32-wide slice
      async_copy_b128(wbf + (long)(nBase + nr) * Kp + kk + kp8, &Bsb[q * 8]);
    }
    async_wait0();
#else
#pragma unroll
    for (int i = 0; i < 2; ++i) {
      int q = tid * 2 + i;
      int nr = q >> 2;
      int kp8 = (q & 3) << 3;
      const __bf16* src = wbf + (long)(nBase + nr) * Kp + kk + kp8;
#pragma unroll
      for (int j = 0; j < 8; ++j) Bsb[q * 8 + j] = src[j];
    }
#endif
    __syncthreads();

    // ---- build fragments from LDS (pure ds_load_b128, no converts) ----
    const v8bf* arow = (const v8bf*)&As[laneM * 32 + grp * 8];
    v8bf a0 = arow[0];   // k = 8g .. 8g+7
    v8bf a1 = arow[2];   // k = 16+8g .. 16+8g+7
    v16bf av;
#pragma unroll
    for (int e = 0; e < 8; ++e) { av[e] = a0[e]; av[e + 8] = a1[e]; }

    const v8bf* brow = (const v8bf*)&Bsb[((wave << 4) + laneM) * 32 + grp * 8];
    v8bf b0 = brow[0];   // k = 8g .. 8g+7
    v8bf b1 = brow[2];   // k = 16+8g .. 16+8g+7
    v16bf bv;
#pragma unroll
    for (int e = 0; e < 8; ++e) { bv[e] = b0[e]; bv[e + 8] = b1[e]; }

    acc = __builtin_amdgcn_wmma_f32_16x16x32_bf16(
        /*neg_a=*/false, av, /*neg_b=*/false, bv,
        /*c_mod=*/(short)0, acc, /*reuse_a=*/false, /*reuse_b=*/false);
    __syncthreads();
  }

  // ---- epilogue: bias + optional leaky-relu, store per C/D layout ----
  float bvv = (n < Cout) ? bias[n] : 0.f;
#pragma unroll
  for (int r = 0; r < 8; ++r) {
    int mg = mTile * 16 + r + 8 * grp;
    if (mg < M && n < Cout) {
      float v = acc[r] + bvv;
      if (leaky) v = (v > 0.f) ? v : 0.1f * v;
      int b2 = mg / HWo;
      int r2 = mg - b2 * HWo;
      out[(long)b2 * oBS + (long)n * HWo + r2] = v;
    }
  }
}

// ---------------------------------------------------------------------------
// Elementwise / gather kernels
// ---------------------------------------------------------------------------
__global__ void warp_kernel(const float* __restrict__ src, long srcBS,
                            const float* __restrict__ flow, long flowBS,
                            float* __restrict__ out, long outBS,
                            int Bn, int C, int H, int W, float sign) {
  long i = (long)blockIdx.x * blockDim.x + threadIdx.x;
  long total = (long)Bn * C * H * W;
  if (i >= total) return;
  int xw = (int)(i % W); long t = i / W;
  int y = (int)(t % H); t /= H;
  int c = (int)(t % C); int b = (int)(t / C);
  long fb = (long)b * flowBS;
  float fx = flow[fb + (long)y * W + xw];
  float fy = flow[fb + (long)H * W + (long)y * W + xw];
  float v = bilerp_zero(src + (long)b * srcBS + (long)c * H * W, H, W,
                        (float)xw + sign * fx, (float)y + sign * fy);
  out[(long)b * outBS + (long)c * H * W + (long)y * W + xw] = v;
}

__global__ void spike_warp_kernel(const float* __restrict__ seq,
                                  const float* __restrict__ flow,
                                  const int* __restrict__ dt,
                                  float* __restrict__ out, int Bn, int H, int W) {
  const int C = 25;
  long i = (long)blockIdx.x * blockDim.x + threadIdx.x;
  long total = (long)Bn * C * H * W;
  if (i >= total) return;
  int xw = (int)(i % W); long t = i / W;
  int y = (int)(t % H); t /= H;
  int c = (int)(t % C); int b = (int)(t / C);
  float f = ((float)c - 12.0f) / (float)dt[0];   // linspace(-12,12,25)[c] / dt
  long fb = (long)b * 2 * H * W;
  float fx = flow[fb + (long)y * W + xw] * f;
  float fy = flow[fb + (long)H * W + (long)y * W + xw] * f;
  out[i] = bilerp_zero(seq + ((long)b * C + c) * H * W, H, W,
                       (float)xw + fx, (float)y + fy);
}

__global__ void corr_kernel(const float* __restrict__ x1, long bs1,
                            const float* __restrict__ x2, long bs2,
                            float* __restrict__ out, long obs,
                            int Bn, int C, int H, int W) {
  long i = (long)blockIdx.x * blockDim.x + threadIdx.x;
  long total = (long)Bn * 81 * H * W;
  if (i >= total) return;
  int xw = (int)(i % W); long t = i / W;
  int y = (int)(t % H); t /= H;
  int d = (int)(t % 81); int b = (int)(t / 81);
  int dy = d / 9 - 4, dx = d % 9 - 4;
  int sy = y + dy, sx = xw + dx;
  float s = 0.f;
  if (sy >= 0 && sy < H && sx >= 0 && sx < W) {
    const float* p1 = x1 + (long)b * bs1 + (long)y * W + xw;
    const float* p2 = x2 + (long)b * bs2 + (long)sy * W + sx;
    long HW = (long)H * W;
    for (int c = 0; c < C; ++c) s += p1[c * HW] * p2[c * HW];
  }
  out[(long)b * obs + (long)d * H * W + (long)y * W + xw] = s / (float)C;
}

// align_corners=True 2x upsample, fused with the flow*2 scaling
__global__ void up2_kernel(const float* __restrict__ in, float* __restrict__ out,
                           int Bn, int C, int H, int W) {
  int Ho = 2 * H, Wo = 2 * W;
  long i = (long)blockIdx.x * blockDim.x + threadIdx.x;
  long total = (long)Bn * C * Ho * Wo;
  if (i >= total) return;
  int X = (int)(i % Wo); long t = i / Wo;
  int Y = (int)(t % Ho); t /= Ho;
  int c = (int)(t % C); int b = (int)(t / C);
  float yc = (float)Y * (float)(H - 1) / (float)(Ho - 1);
  float xc = (float)X * (float)(W - 1) / (float)(Wo - 1);
  int y0 = (int)yc, x0 = (int)xc;
  int y1 = (y0 + 1 < H) ? y0 + 1 : H - 1;
  int x1 = (x0 + 1 < W) ? x0 + 1 : W - 1;
  float wy = yc - (float)y0, wx = xc - (float)x0;
  const float* p = in + ((long)b * C + c) * H * W;
  float v = (p[(long)y0 * W + x0] * (1.f - wx) + p[(long)y0 * W + x1] * wx) * (1.f - wy) +
            (p[(long)y1 * W + x0] * (1.f - wx) + p[(long)y1 * W + x1] * wx) * wy;
  out[((long)b * C + c) * Ho * Wo + (long)Y * Wo + X] = 2.f * v;
}

__global__ void copy_chan_kernel(const float* __restrict__ src, long sBS,
                                 float* __restrict__ dst, long dBS,
                                 int Bn, int C, long HW) {
  long i = (long)blockIdx.x * blockDim.x + threadIdx.x;
  long total = (long)Bn * C * HW;
  if (i >= total) return;
  long pix = i % HW; long t = i / HW;
  int c = (int)(t % C); int b = (int)(t / C);
  dst[(long)b * dBS + (long)c * HW + pix] = src[(long)b * sBS + (long)c * HW + pix];
}

__global__ void add_kernel(float* __restrict__ a, const float* __restrict__ b, long n) {
  long i = (long)blockIdx.x * blockDim.x + threadIdx.x;
  if (i < n) a[i] += b[i];
}

__global__ void copy_kernel(float* __restrict__ dst, const float* __restrict__ src, long n) {
  long i = (long)blockIdx.x * blockDim.x + threadIdx.x;
  if (i < n) dst[i] = src[i];
}

__global__ void zero_kernel(float* __restrict__ dst, long n) {
  long i = (long)blockIdx.x * blockDim.x + threadIdx.x;
  if (i < n) dst[i] = 0.f;
}

__global__ void mean_abs_kernel(const float* __restrict__ x, float* __restrict__ out,
                                int Bn, int C, long HW) {
  long i = (long)blockIdx.x * blockDim.x + threadIdx.x;
  long total = (long)Bn * HW;
  if (i >= total) return;
  long pix = i % HW; int b = (int)(i / HW);
  const float* p = x + (long)b * C * HW + pix;
  float s = 0.f;
  for (int c = 0; c < C; ++c) s += fabsf(p[c * HW]);
  out[i] = s / (float)C;
}

// ---------------------------------------------------------------------------
// Host-side helpers
// ---------------------------------------------------------------------------
static inline dim3 ew_grid(long n) { return dim3((unsigned)((n + 255) / 256)); }

static inline void launch_conv(hipStream_t s, const float* x, long xBS,
                               const __bf16* wbf, int Kp, const float* b,
                               float* out, long oBS,
                               int Bn, int Cin, int Hin, int Win,
                               int Cout, int Hout, int Wout,
                               int ks, int stride, int leaky) {
  int M = Bn * Hout * Wout;
  dim3 grid((M + 15) / 16, (Cout + 63) / 64);
  if (ks == 3)
    conv_wmma_kernel<3><<<grid, 128, 0, s>>>(x, xBS, wbf, Kp, b, out, oBS, Bn, Cin, Hin, Win,
                                             Cout, Hout, Wout, stride, leaky);
  else
    conv_wmma_kernel<1><<<grid, 128, 0, s>>>(x, xBS, wbf, Kp, b, out, oBS, Bn, Cin, Hin, Win,
                                             Cout, Hout, Wout, stride, leaky);
}

// ---------------------------------------------------------------------------
extern "C" void kernel_launch(void* const* d_in, const int* in_sizes, int n_in,
                              void* d_out, int out_size, void* d_ws, size_t ws_size,
                              hipStream_t stream) {
  (void)in_sizes; (void)n_in; (void)out_size; (void)ws_size;
  // ---- inputs ----
  const float* seq1    = (const float*)d_in[0];   // (2,25,256,320)
  const float* seq2    = (const float*)d_in[1];
  const float* flow_in = (const float*)d_in[2];   // (2,2,256,320)
  const int*   dt      = (const int*)d_in[3];
  // params (dict insertion order)
  const float* s1_w = (const float*)d_in[4];  const float* s1_b = (const float*)d_in[5];
  const float* s2_w = (const float*)d_in[6];  const float* s2_b = (const float*)d_in[7];
  const float* encw[4][2]; const float* encb[4][2];
  for (int l = 0; l < 4; ++l) {
    encw[l][0] = (const float*)d_in[8 + 4 * l + 0];
    encb[l][0] = (const float*)d_in[8 + 4 * l + 1];
    encw[l][1] = (const float*)d_in[8 + 4 * l + 2];
    encb[l][1] = (const float*)d_in[8 + 4 * l + 3];
  }
  const float* c1w[4]; const float* c1b[4];
  for (int l = 0; l < 4; ++l) {
    c1w[l] = (const float*)d_in[24 + 2 * l];
    c1b[l] = (const float*)d_in[25 + 2 * l];
  }
  const float* fe_w[4]; const float* fe_b[4];
  for (int l = 0; l < 4; ++l) {
    fe_w[l] = (const float*)d_in[32 + 2 * l];
    fe_b[l] = (const float*)d_in[33 + 2 * l];
  }
  float* out = (float*)d_out;

  const int Bn = 2, H = 256, W = 320;
  const long HW = (long)H * W;  // 81920

  // ---- workspace layout (floats) ----
  float* ws = (float*)d_ws;
  size_t off = 0;
  auto alloc = [&](size_t n) { float* p = ws + off; off += n; return p; };
  float* flowW   = alloc((size_t)Bn * 2 * HW);       // warped flow for seq2 rep
  float* warpbuf = alloc((size_t)Bn * 25 * HW);      // warped 25-ch sequence
  float* tmp     = alloc((size_t)Bn * 32 * HW);      // conv temp (largest use)
  float* x1r     = alloc((size_t)Bn * 32 * HW);
  float* x2r     = alloc((size_t)Bn * 32 * HW);
  float* f1[4], *f2[4];
  const int eh[4] = {256, 128, 64, 32}, ew_[4] = {320, 160, 80, 40};
  const int ec[4] = {32, 64, 96, 128};
  for (int l = 0; l < 4; ++l) f1[l] = alloc((size_t)Bn * ec[l] * eh[l] * ew_[l]);
  for (int l = 0; l < 4; ++l) f2[l] = alloc((size_t)Bn * ec[l] * eh[l] * ew_[l]);
  float* flowA   = alloc((size_t)Bn * 2 * HW);
  float* flowB   = alloc((size_t)Bn * 2 * HW);
  float* flowres = alloc((size_t)Bn * 2 * HW);
  float* x2warp  = alloc((size_t)Bn * 32 * HW);
  float* catbuf  = alloc((size_t)Bn * 307 * HW);     // dense-net concat buffer

  // ---- bf16 weight shadows (padded [ceil64(Cout)][ceil32(K)]) ----
  struct WSrc { const float* src; int Cout; int K; };
  const WSrc wl[18] = {
      {s1_w, 32, 225}, {s2_w, 32, 288},
      {encw[0][0], 32, 288}, {encw[0][1], 32, 288},
      {encw[1][0], 64, 288}, {encw[1][1], 64, 576},
      {encw[2][0], 96, 576}, {encw[2][1], 96, 864},
      {encw[3][0], 128, 864}, {encw[3][1], 128, 1152},
      {c1w[0], 32, 128}, {c1w[1], 32, 96}, {c1w[2], 32, 64}, {c1w[3], 32, 32},
      {fe_w[0], 96, 1035}, {fe_w[1], 64, 1899}, {fe_w[2], 32, 2475}, {fe_w[3], 2, 2763}};
  const __bf16* wbfp[18];
  int wKp[18];
  {
    __bf16* base = (__bf16*)alloc(1);  // placeholder to get a 4B-aligned cursor
    size_t boff = 0;                   // in bf16 elements, kept 8-aligned (16B)
    for (int i = 0; i < 18; ++i) {
      int Kp = (wl[i].K + 31) & ~31;
      int Np = (wl[i].Cout + 63) & ~63;
      long total = (long)Np * Kp;
      __bf16* dst = base + boff;
      w2bf_kernel<<<ew_grid(total), 256, 0, stream>>>(wl[i].src, dst, wl[i].Cout,
                                                      wl[i].K, Kp, total);
      wbfp[i] = dst;
      wKp[i] = Kp;
      boff += (size_t)((total + 7) & ~7L);
    }
    off += (boff + 1) / 2;  // account bf16 region in float units
  }

  // ---- spike representation ----
  // flowW = flow_warp(flow_in, -flow_in)
  {
    long n = (long)Bn * 2 * HW;
    warp_kernel<<<ew_grid(n), 256, 0, stream>>>(flow_in, 2 * HW, flow_in, 2 * HW,
                                                flowW, 2 * HW, Bn, 2, H, W, -1.f);
  }
  // x1r = conv(conv(spike_warp(seq1, flow_in)))
  {
    long n = (long)Bn * 25 * HW;
    spike_warp_kernel<<<ew_grid(n), 256, 0, stream>>>(seq1, flow_in, dt, warpbuf, Bn, H, W);
  }
  launch_conv(stream, warpbuf, 25 * HW, wbfp[0], wKp[0], s1_b, tmp, 32 * HW,
              Bn, 25, H, W, 32, H, W, 3, 1, 1);
  launch_conv(stream, tmp, 32 * HW, wbfp[1], wKp[1], s2_b, x1r, 32 * HW,
              Bn, 32, H, W, 32, H, W, 3, 1, 1);
  // x2r
  {
    long n = (long)Bn * 25 * HW;
    spike_warp_kernel<<<ew_grid(n), 256, 0, stream>>>(seq2, flowW, dt, warpbuf, Bn, H, W);
  }
  launch_conv(stream, warpbuf, 25 * HW, wbfp[0], wKp[0], s1_b, tmp, 32 * HW,
              Bn, 25, H, W, 32, H, W, 3, 1, 1);
  launch_conv(stream, tmp, 32 * HW, wbfp[1], wKp[1], s2_b, x2r, 32 * HW,
              Bn, 32, H, W, 32, H, W, 3, 1, 1);

  // ---- feature encoders ----
  const int ci_in[4] = {32, 32, 64, 96};
  for (int which = 0; which < 2; ++which) {
    const float* xin = which ? x2r : x1r;
    float** f = which ? f2 : f1;
    for (int l = 0; l < 4; ++l) {
      int s = (l == 0) ? 1 : 2;
      int Hin = (l == 0) ? H : eh[l - 1];
      int Win = (l == 0) ? W : ew_[l - 1];
      long inBS = (long)ci_in[l] * Hin * Win;
      long obs = (long)ec[l] * eh[l] * ew_[l];
      const float* src = (l == 0) ? xin : f[l - 1];
      launch_conv(stream, src, inBS, wbfp[2 + 2 * l], wKp[2 + 2 * l], encb[l][0], tmp, obs,
                  Bn, ci_in[l], Hin, Win, ec[l], eh[l], ew_[l], 3, s, 1);
      launch_conv(stream, tmp, obs, wbfp[3 + 2 * l], wKp[3 + 2 * l], encb[l][1], f[l], obs,
                  Bn, ec[l], eh[l], ew_[l], ec[l], eh[l], ew_[l], 3, 1, 1);
    }
  }

  // ---- pyramid flow estimation (coarse -> fine) ----
  const int lh[4] = {32, 64, 128, 256}, lw[4] = {40, 80, 160, 320};
  const int Cf[4] = {128, 96, 64, 32};
  float* x1f[4] = {f1[3], f1[2], f1[1], f1[0]};
  float* x2f[4] = {f2[3], f2[2], f2[1], f2[0]};
  const long outOff[4] = {430080, 409600, 327680, 0};
  const int feCin[4] = {115, 211, 275, 307};
  const int feCout[4] = {96, 64, 32, 2};
  // catbuf channel layout: [fe3(32) | fe2(64) | fe1(96) | corr(81) | x1_1x1(32) | flow(2)]
  // offsets:                 0        32        96        192         273          305
  const int feInOff[4] = {192, 96, 32, 0};
  const int feOutOff[4] = {96, 32, 0, -1};  // fe4 -> flowres

  float* flowCur = flowA;
  float* flowNext = flowB;
  for (int l = 0; l < 4; ++l) {
    int h = lh[l], w = lw[l];
    long hw = (long)h * w;
    long catBS = 307L * hw;
    long fBS = (long)Cf[l] * hw;
    const float* x2use;
    if (l == 0) {
      zero_kernel<<<ew_grid((long)Bn * 2 * hw), 256, 0, stream>>>(flowCur, (long)Bn * 2 * hw);
      x2use = x2f[0];
    } else {
      // flow = up2(flow*2); x2_warp = flow_warp(x2, flow)
      long n = (long)Bn * 2 * hw;
      up2_kernel<<<ew_grid(n), 256, 0, stream>>>(flowCur, flowNext, Bn, 2, h / 2, w / 2);
      float* t = flowCur; flowCur = flowNext; flowNext = t;
      long nw = (long)Bn * Cf[l] * hw;
      warp_kernel<<<ew_grid(nw), 256, 0, stream>>>(x2f[l], fBS, flowCur, 2 * hw,
                                                   x2warp, fBS, Bn, Cf[l], h, w, 1.f);
      x2use = x2warp;
    }
    // corr -> catbuf channels [192:273]
    {
      long n = (long)Bn * 81 * hw;
      corr_kernel<<<ew_grid(n), 256, 0, stream>>>(x1f[l], fBS, x2use, fBS,
                                                  catbuf + 192 * hw, catBS, Bn, Cf[l], h, w);
    }
    // 1x1 conv -> catbuf channels [273:305]
    launch_conv(stream, x1f[l], fBS, wbfp[10 + l], wKp[10 + l], c1b[l],
                catbuf + 273 * hw, catBS, Bn, Cf[l], h, w, 32, h, w, 1, 1, 0);
    // flow -> catbuf channels [305:307]
    {
      long n = (long)Bn * 2 * hw;
      copy_chan_kernel<<<ew_grid(n), 256, 0, stream>>>(flowCur, 2 * hw,
                                                       catbuf + 305 * hw, catBS, Bn, 2, hw);
    }
    // dense-net flow estimator
    for (int k = 0; k < 4; ++k) {
      const float* src = catbuf + (long)feInOff[k] * hw;
      float* dst; long dBS; int leak;
      if (k < 3) { dst = catbuf + (long)feOutOff[k] * hw; dBS = catBS; leak = 1; }
      else       { dst = flowres;                          dBS = 2 * hw; leak = 0; }
      launch_conv(stream, src, catBS, wbfp[14 + k], wKp[14 + k], fe_b[k], dst, dBS,
                  Bn, feCin[k], h, w, feCout[k], h, w, 3, 1, leak);
    }
    // flow += flow_res; write to output slot
    {
      long n = (long)Bn * 2 * hw;
      add_kernel<<<ew_grid(n), 256, 0, stream>>>(flowCur, flowres, n);
      copy_kernel<<<ew_grid(n), 256, 0, stream>>>(out + outOff[l], flowCur, n);
    }
  }

  // ---- x1m / x2m ----
  {
    long n = (long)Bn * HW;
    mean_abs_kernel<<<ew_grid(n), 256, 0, stream>>>(x1r, out + 435200, Bn, 32, HW);
    mean_abs_kernel<<<ew_grid(n), 256, 0, stream>>>(x2r, out + 599040, Bn, 32, HW);
  }
}
